// LSTM_with_GCC_28570122453319
// MI455X (gfx1250) — matrix operand order repaired and checked by reading, not compiled, not run on hardware
//
#include <hip/hip_runtime.h>
#include <hip/hip_bf16.h>
#include <stdint.h>

// ---------------------------------------------------------------------------
// LSTM with gated-context contributions, CDNA5 (gfx1250) WMMA implementation.
// B=64, T=128, IN=1024, H=1024, F0=1024, F1=2048. fp32 in/out, bf16 WMMA math.
// Fused single-pass gx GEMM (no RMW, no epilogue branches), fused step GEMM +
// LSTM cell pointwise (one launch/step), async global->LDS staging.
// ---------------------------------------------------------------------------

typedef __bf16 bf16;
typedef __bf16 v8bf  __attribute__((ext_vector_type(8)));
typedef __bf16 v16bf __attribute__((ext_vector_type(16)));
typedef float  v8f   __attribute__((ext_vector_type(8)));

#define LDA_PITCH 40   // bf16 elems per LDS row (80B: 16B-aligned, bank-friendly)

static constexpr int Bsz = 64;
static constexpr int T   = 128;
static constexpr int IN  = 1024;
static constexpr int H   = 1024;
static constexpr int G4  = 4 * H;   // 4096
static constexpr int G3  = 3 * H;   // 3072
static constexpr int F1  = 2048;

// ---------------------------------------------------------------------------
// Global -> LDS 16B copy. Prefer CDNA5 GLOBAL_LOAD_ASYNC_TO_LDS_B128
// (ASYNCcnt-tracked) when the toolchain exposes it; else load+ds_store.
// Builtin signature (per hipcc diagnostic): (v4i AS1*, v4i AS3*, Iint, Iint).
// ---------------------------------------------------------------------------
#if defined(__AMDGCN__) && __has_builtin(__builtin_amdgcn_global_load_async_to_lds_b128) && \
    __has_builtin(__builtin_amdgcn_s_wait_asynccnt)
#define HAVE_ASYNC_LDS 1
#else
#define HAVE_ASYNC_LDS 0
#endif

#if HAVE_ASYNC_LDS
typedef int v4i __attribute__((ext_vector_type(4)));
typedef __attribute__((address_space(1))) v4i* glb_v4i_p;
typedef __attribute__((address_space(3))) v4i* lds_v4i_p;
#endif

__device__ __forceinline__ void g2l_copy16(const bf16* __restrict__ g, bf16* l) {
#if HAVE_ASYNC_LDS
  // Generic LDS pointer low 32 bits == LDS offset; AS1 numeric == generic.
  __builtin_amdgcn_global_load_async_to_lds_b128(
      (glb_v4i_p)(uintptr_t)g, (lds_v4i_p)(uint32_t)(uintptr_t)l, 0, 0);
#else
  *(v8bf*)l = *(const v8bf*)g;
#endif
}

__device__ __forceinline__ void wait_copies() {
#if HAVE_ASYNC_LDS
  __builtin_amdgcn_s_wait_asynccnt(0);
#endif
}

// ---------------------------------------------------------------------------
// Elementwise fp32 -> bf16 cast (grid-stride)
// ---------------------------------------------------------------------------
__global__ void cast_f32_bf16(const float* __restrict__ src, bf16* __restrict__ dst, int n) {
  for (int i = blockIdx.x * blockDim.x + threadIdx.x; i < n; i += gridDim.x * blockDim.x)
    dst[i] = (bf16)src[i];
}

// ---------------------------------------------------------------------------
// Pack weight W [N][K] (row-major fp32) into WMMA-B-fragment-major bf16.
// Fragment f = kt*(N/16)+nt is the 32x16 tile B[kt*32..+31][nt*16..+15] stored
// in per-lane order: lane l (n = nt*16+(l&15)) holds K = kt*32+(l>>4)*16+j,
// j=0..15 contiguous at P[f*512 + l*16 + j].   (B[k][n] = W[n][k])
// ---------------------------------------------------------------------------
__global__ void pack_w_frag(const float* __restrict__ W, bf16* __restrict__ P, int N, int K) {
  const int total = N * K;
  const int nfr = N >> 4;
  for (int idx = blockIdx.x * blockDim.x + threadIdx.x; idx < total;
       idx += gridDim.x * blockDim.x) {
    int f = idx >> 9;
    int l = (idx >> 4) & 31;
    int j = idx & 15;
    int nt = f % nfr;
    int kt = f / nfr;
    int n = nt * 16 + (l & 15);
    int k = kt * 32 + (l >> 4) * 16 + j;
    P[idx] = (bf16)W[(size_t)n * K + k];
  }
}

// ---------------------------------------------------------------------------
// Combined bias vector over the 4H gate axis.
// ---------------------------------------------------------------------------
__global__ void make_bvec(const float* __restrict__ bih, const float* __restrict__ bhh,
                          const float* __restrict__ bf0, const float* __restrict__ bf1,
                          float* __restrict__ bvec) {
  int c = blockIdx.x * blockDim.x + threadIdx.x;
  if (c >= G4) return;
  float v = bih[c] + bhh[c];
  if (c < 2 * H)        v += bf0[c] + bf1[c];           // i, f gates
  else if (c >= 3 * H)  v += bf0[c - H] + bf1[c - H];   // o gate
  bvec[c] = v;
}

// ---------------------------------------------------------------------------
// Fragment loads from LDS (same contiguous k order in A and B).
// A (16x32): lane l holds row m=l&15, K = (l>=16?16:0) + 0..15 contiguous.
// B (32x16): fragment-packed, lane l reads 32B at frag*512 + l*16.
// ---------------------------------------------------------------------------
__device__ __forceinline__ v16bf load_a_frag(const bf16* __restrict__ As, int mloc, int lane) {
  const int k0 = (lane >> 4) * 16;
  v8bf lo = *(const v8bf*)(&As[mloc * LDA_PITCH + k0]);
  v8bf hi = *(const v8bf*)(&As[mloc * LDA_PITCH + k0 + 8]);
  return __builtin_shufflevector(lo, hi, 0, 1, 2, 3, 4, 5, 6, 7, 8, 9, 10, 11, 12, 13, 14, 15);
}

__device__ __forceinline__ v16bf load_b_frag(const bf16* __restrict__ Bs, int frag, int lane) {
  const bf16* p = &Bs[frag * 512 + lane * 16];
  v8bf lo = *(const v8bf*)(p);
  v8bf hi = *(const v8bf*)(p + 8);
  return __builtin_shufflevector(lo, hi, 0, 1, 2, 3, 4, 5, 6, 7, 8, 9, 10, 11, 12, 13, 14, 15);
}

// ---------------------------------------------------------------------------
// One accumulation phase of the fused projection GEMM over a 128x128 C-tile.
// A: bf16 [8192][K] row-major; Bp: fragment-packed with nfr = N/16 fragments
// per 32-row K-tile; fcol = first fragment column of this C-tile.
// ---------------------------------------------------------------------------
__device__ __forceinline__ void mma_phase(const bf16* __restrict__ A, int K,
                                          const bf16* __restrict__ Bp, int nfr, int fcol,
                                          bf16* As, bf16* Bs, int m0, int tid, int lane,
                                          int waveM, int waveN, v8f (&acc)[2][4]) {
  for (int kb = 0; kb < K; kb += 32) {
    // A tile 128x32: 512 x 16B chunks, 2/thread
    for (int it = 0; it < 2; ++it) {
      int c = it * 256 + tid;
      int row = c >> 2, ch = c & 3;
      g2l_copy16(A + (size_t)(m0 + row) * K + kb + ch * 8, &As[row * LDA_PITCH + ch * 8]);
    }
    // B tile: 8 fragments x 512 bf16 contiguous
    const size_t fbase = ((size_t)(kb >> 5) * nfr + fcol) * 512;
    for (int it = 0; it < 2; ++it) {
      int c = it * 256 + tid;
      g2l_copy16(Bp + fbase + (size_t)c * 8, &Bs[c * 8]);
    }
    wait_copies();
    __syncthreads();

    v16bf af[2], bfr[4];
    for (int mi = 0; mi < 2; ++mi)
      af[mi] = load_a_frag(As, waveM * 32 + mi * 16 + (lane & 15), lane);
    for (int ni = 0; ni < 4; ++ni)
      bfr[ni] = load_b_frag(Bs, waveN * 4 + ni, lane);
    for (int mi = 0; mi < 2; ++mi)
      for (int ni = 0; ni < 4; ++ni)
        acc[mi][ni] = __builtin_amdgcn_wmma_f32_16x16x32_bf16(
            false, af[mi], false, bfr[ni], (short)0, acc[mi][ni], false, false);
    __syncthreads();
  }
}

// ---------------------------------------------------------------------------
// Fused time-parallel projection:
//   gx[m][n] = bvec[n] + (X@Wih^T)[m][n] + [aux0@Wfh0^T + aux1@Wfh1^T](mapped)
// Each 128-wide column tile lies in exactly one gate region, so the aux
// column map is a per-tile constant; g-gate tiles skip the aux phases.
// Single C write per element: no read-modify-write, no branches per store.
// ---------------------------------------------------------------------------
__global__ __launch_bounds__(256)
void gemm_gx(const bf16* __restrict__ xbf, const bf16* __restrict__ a0bf,
             const bf16* __restrict__ a1bf, const bf16* __restrict__ wihP,
             const bf16* __restrict__ wfh0P, const bf16* __restrict__ wfh1P,
             const float* __restrict__ bvec, float* __restrict__ gx) {
  __shared__ __align__(16) bf16 As[128 * LDA_PITCH];
  __shared__ __align__(16) bf16 Bs[32 * 128];

  const int tid = threadIdx.x;
  const int lane = tid & 31;
  const int wave = tid >> 5;      // wave32
  const int waveM = wave >> 1;    // 0..3 : 32 rows
  const int waveN = wave & 1;     // 0..1 : 64 cols
  const int n0 = blockIdx.x * 128;
  const int m0 = blockIdx.y * 128;

  v8f acc[2][4];
  for (int i = 0; i < 2; ++i)
    for (int j = 0; j < 4; ++j) { v8f z = {}; acc[i][j] = z; }

  mma_phase(xbf, IN, wihP, G4 >> 4, n0 >> 4, As, Bs, m0, tid, lane, waveM, waveN, acc);

  const bool has_aux = !(n0 >= 2 * H && n0 < 3 * H);   // cellgate gets no aux
  if (has_aux) {
    const int an0 = (n0 >= 3 * H) ? n0 - H : n0;       // gate col -> aux col
    mma_phase(a0bf, H,  wfh0P, G3 >> 4, an0 >> 4, As, Bs, m0, tid, lane, waveM, waveN, acc);
    mma_phase(a1bf, F1, wfh1P, G3 >> 4, an0 >> 4, As, Bs, m0, tid, lane, waveM, waveN, acc);
  }

  // C/D layout: lane -> N, (lane>>4)*8 + vgpr -> M.  Single store per element.
  for (int mi = 0; mi < 2; ++mi) {
    int mbase = m0 + waveM * 32 + mi * 16 + (lane >> 4) * 8;
    for (int ni = 0; ni < 4; ++ni) {
      int n = n0 + waveN * 64 + ni * 16 + (lane & 15);
      float bv = bvec[n];
      for (int r = 0; r < 8; ++r)
        gx[(size_t)(mbase + r) * G4 + n] = bv + acc[mi][ni][r];
    }
  }
}

// ---------------------------------------------------------------------------
// Fused recurrent step: GEMM + LSTM cell.
// Each WG owns hc-slice [hc0, hc0+32) and computes all 4 gate columns for it
// (cols g*H + hc), so the cell update runs on WMMA accumulators in-register.
// h state is ping-ponged (Hin/Hout) since every WG reads the full h vector.
//   gates[b][g*H+hc] = gx[b][t][g*H+hc] + sum_k h[b][k] * Whh[g*H+hc][k]
// ---------------------------------------------------------------------------
__global__ __launch_bounds__(256)
void lstm_step(const bf16* __restrict__ Hin, bf16* __restrict__ Hout,
               const bf16* __restrict__ whhP, const float* __restrict__ GX,
               float* __restrict__ out, float* __restrict__ hT,
               float* __restrict__ cT, int t) {
  __shared__ __align__(16) bf16 As[64 * LDA_PITCH];
  __shared__ __align__(16) bf16 Bs[8 * 512];

  const int tid = threadIdx.x;
  const int lane = tid & 31;
  const int wave = tid >> 5;
  const int waveM = wave >> 1;   // 0..3 : 16 batch rows each
  const int waveN = wave & 1;    // 0..1 : 16 hc cols each
  const int hc0 = blockIdx.x * 32;

  v8f acc[4];
  for (int g = 0; g < 4; ++g) { v8f z = {}; acc[g] = z; }

  for (int kb = 0; kb < H; kb += 32) {
    {  // A tile 64x32 (h state): 256 chunks, 1/thread
      int row = tid >> 2, ch = tid & 3;
      g2l_copy16(Hin + (size_t)row * H + kb + ch * 8, &As[row * LDA_PITCH + ch * 8]);
    }
    // B: 8 fragments (4 gates x 2 hc-fragments), 512 chunks, 2/thread
    for (int it = 0; it < 2; ++it) {
      int c = it * 256 + tid;
      int fl = c >> 6;                 // 0..7 local fragment
      int g = fl >> 1, fi = fl & 1;
      size_t f = (size_t)(kb >> 5) * (G4 >> 4) + (size_t)g * (H >> 4) + (hc0 >> 4) + fi;
      g2l_copy16(whhP + f * 512 + (size_t)(c & 63) * 8, &Bs[fl * 512 + (c & 63) * 8]);
    }
    wait_copies();
    __syncthreads();

    v16bf af = load_a_frag(As, waveM * 16 + (lane & 15), lane);
    for (int g = 0; g < 4; ++g) {
      v16bf bfr = load_b_frag(Bs, g * 2 + waveN, lane);
      acc[g] = __builtin_amdgcn_wmma_f32_16x16x32_bf16(
          false, af, false, bfr, (short)0, acc[g], false, false);
    }
    __syncthreads();
  }

  // In-register LSTM cell update for this wave's 16x16 (b, hc) tile.
  const int hc = hc0 + waveN * 16 + (lane & 15);
  const int mbase = waveM * 16 + (lane >> 4) * 8;
  for (int r = 0; r < 8; ++r) {
    int b = mbase + r;
    size_t gxb = ((size_t)b * T + t) * (size_t)G4 + hc;
    float gi = 1.0f / (1.0f + __expf(-(acc[0][r] + GX[gxb])));
    float gf = 1.0f / (1.0f + __expf(-(acc[1][r] + GX[gxb + H])));
    float gg = tanhf(acc[2][r] + GX[gxb + 2 * H]);
    float go = 1.0f / (1.0f + __expf(-(acc[3][r] + GX[gxb + 3 * H])));
    int ci = b * H + hc;
    float cy = gf * cT[ci] + gi * gg;
    float hy = go * tanhf(cy);
    cT[ci] = cy;
    hT[ci] = hy;                                   // last step leaves hT
    out[((size_t)b * T + t) * (size_t)H + hc] = hy;
    Hout[ci] = (bf16)hy;
  }
}

__global__ void init_state(float* __restrict__ hT, float* __restrict__ cT,
                           bf16* __restrict__ Hbf) {
  int idx = blockIdx.x * blockDim.x + threadIdx.x;
  if (idx >= Bsz * H) return;
  hT[idx] = 0.0f;
  cT[idx] = 0.0f;
  Hbf[idx] = (bf16)0.0f;
}

// ---------------------------------------------------------------------------
// Host-side orchestration
// ---------------------------------------------------------------------------
extern "C" void kernel_launch(void* const* d_in, const int* in_sizes, int n_in,
                              void* d_out, int out_size, void* d_ws, size_t ws_size,
                              hipStream_t stream) {
  const float* input_feats = (const float*)d_in[0];   // [B,T,IN]
  const float* aux0        = (const float*)d_in[1];   // [B,T,F0]
  const float* aux1        = (const float*)d_in[2];   // [B,T,F1]
  const float* weight_ih   = (const float*)d_in[3];   // [4H,IN]
  const float* weight_hh   = (const float*)d_in[4];   // [4H,H]
  const float* weight_fh0  = (const float*)d_in[5];   // [3H,F0]
  const float* weight_fh1  = (const float*)d_in[6];   // [3H,F1]
  const float* bias_ih     = (const float*)d_in[7];
  const float* bias_hh     = (const float*)d_in[8];
  const float* bias_fh0    = (const float*)d_in[9];
  const float* bias_fh1    = (const float*)d_in[10];

  float* out = (float*)d_out;                    // [B,T,H]
  float* hT  = out + (size_t)Bsz * T * H;        // [B,H]
  float* cT  = hT + (size_t)Bsz * H;             // [B,H]

  char* ws = (char*)d_ws;
  size_t off = 0;
  auto carve = [&](size_t bytes) { void* p = ws + off; off = (off + bytes + 255) & ~(size_t)255; return p; };
  const size_t MR = (size_t)Bsz * T;             // 8192 rows
  bf16*  x_bf   = (bf16*)carve(MR * IN * 2);
  bf16*  a0_bf  = (bf16*)carve(MR * H * 2);
  bf16*  a1_bf  = (bf16*)carve(MR * F1 * 2);
  bf16*  wihP   = (bf16*)carve((size_t)G4 * IN * 2);
  bf16*  wfh0P  = (bf16*)carve((size_t)G3 * H * 2);
  bf16*  wfh1P  = (bf16*)carve((size_t)G3 * F1 * 2);
  bf16*  whhP   = (bf16*)carve((size_t)G4 * H * 2);
  float* gx     = (float*)carve(MR * G4 * 4);    // 128 MB, L2-resident
  float* bvec   = (float*)carve((size_t)G4 * 4);
  bf16*  h0     = (bf16*)carve((size_t)Bsz * H * 2);   // h ping-pong
  bf16*  h1     = (bf16*)carve((size_t)Bsz * H * 2);
  (void)ws_size; (void)in_sizes; (void)n_in; (void)out_size;

  // 1) casts
  cast_f32_bf16<<<4096, 256, 0, stream>>>(input_feats, x_bf, (int)(MR * IN));
  cast_f32_bf16<<<4096, 256, 0, stream>>>(aux0, a0_bf, (int)(MR * H));
  cast_f32_bf16<<<4096, 256, 0, stream>>>(aux1, a1_bf, (int)(MR * F1));

  // 2) fragment-pack weights
  pack_w_frag<<<4096, 256, 0, stream>>>(weight_ih,  wihP,  G4, IN);
  pack_w_frag<<<4096, 256, 0, stream>>>(weight_fh0, wfh0P, G3, H);
  pack_w_frag<<<4096, 256, 0, stream>>>(weight_fh1, wfh1P, G3, F1);
  pack_w_frag<<<4096, 256, 0, stream>>>(weight_hh,  whhP,  G4, H);

  // 3) bias vector
  make_bvec<<<G4 / 256, 256, 0, stream>>>(bias_ih, bias_hh, bias_fh0, bias_fh1, bvec);

  // 4) fused time-parallel projections into gx [B*T, 4H] (single pass)
  gemm_gx<<<dim3(G4 / 128, MR / 128), 256, 0, stream>>>(
      x_bf, a0_bf, a1_bf, wihP, wfh0P, wfh1P, bvec, gx);

  // 5) recurrence: one fused kernel per step, ping-ponged h state
  init_state<<<(Bsz * H) / 256, 256, 0, stream>>>(hT, cT, h0);
  for (int t = 0; t < T; ++t) {
    const bf16* Hin = (t & 1) ? h1 : h0;
    bf16* Hout      = (t & 1) ? h0 : h1;
    lstm_step<<<H / 32, 256, 0, stream>>>(Hin, Hout, whhP, gx, out, hT, cT, t);
  }
}
